// SceneGraphModel_72980084293699
// MI455X (gfx1250) — matrix operand order repaired and checked by reading, not compiled, not run on hardware
//
#include <hip/hip_runtime.h>

typedef __attribute__((ext_vector_type(16))) _Float16 v16h;
typedef __attribute__((ext_vector_type(8)))  float    v8f;
typedef __attribute__((ext_vector_type(4)))  unsigned int u32x4;

union U16h { v16h v; u32x4 q[2]; };

// ---------------- utility kernels ----------------

__global__ void k_zero_f32(float* __restrict__ p, long n) {
  long i = (long)blockIdx.x * 256 + threadIdx.x;
  if (i < n) p[i] = 0.0f;
}

__global__ void k_copy_y(const int* __restrict__ y, int* __restrict__ outy, int E) {
  int i = blockIdx.x * 256 + threadIdx.x;
  if (i < E) outy[i] = y[i];
}

// Convert object_fc weights (fp32, [Nout,K] row-major, out = in @ W^T) into f16
// WMMA B-fragments.  Fragment t = nt*KT + kt covers K rows kt*32..+31 and
// N cols nt*16..+15.  B (32x16) in contiguous k order, same as A: lane l<16 holds
// K = kt*32 + h (h = 0..15), lane l>=16 holds K = kt*32+16+h; N = nt*16 + (l&15).
// Stored as frag[t][lane][h] so each lane loads 16 contiguous halves (32 B).
__global__ void k_prep_w(const float* __restrict__ w1, const float* __restrict__ w2,
                         const float* __restrict__ w3, const float* __restrict__ w4,
                         _Float16* __restrict__ frags) {
  int i = blockIdx.x * 256 + threadIdx.x;
  if (i >= 43520) return;
  const float* W; int K, KT, base;
  if      (i < 32768) { W = w1; K = 256; KT = 8; base = 0;     }  // 256->128
  else if (i < 40960) { W = w2; K = 128; KT = 4; base = 32768; }  // 128->64
  else if (i < 43008) { W = w3; K = 64;  KT = 2; base = 40960; }  // 64->32
  else                { W = w4; K = 32;  KT = 1; base = 43008; }  // 32->16
  int r = i - base;
  int t = r >> 9, lane = (r >> 4) & 31, h = r & 15;
  int kt = t & (KT - 1), nt = t / KT;
  int n = nt * 16 + (lane & 15);
  int k = kt * 32 + ((lane >> 4) << 4) + h;
  frags[i] = (_Float16)W[n * K + k];
}

// ---------------- SAGE layers ----------------

__global__ void k_degree(const int* __restrict__ dst, float* __restrict__ cnt, int E) {
  int i = blockIdx.x * 256 + threadIdx.x;
  if (i < E) atomicAdd(&cnt[dst[i]], 1.0f);
}

// s[dst[e]][f] += x[src[e]][f] ; F = 1<<logF features
__global__ void k_scatter(const int* __restrict__ src, const int* __restrict__ dst,
                          const float* __restrict__ xin, float* __restrict__ s,
                          int logF, int E) {
  long i = (long)blockIdx.x * 256 + threadIdx.x;
  int e = (int)(i >> logF);
  int f = (int)(i & ((1 << logF) - 1));
  if (e < E)
    atomicAdd(&s[(size_t)dst[e] * (1 << logF) + f], xin[(size_t)src[e] * (1 << logF) + f]);
}

// h1[node][o] = relu(b[o] + mean . w_l[o,:] + x[node,:] . w_r[o,:]),  16 -> 64
__global__ void k_node1(const float* __restrict__ x, const float* __restrict__ s1,
                        const float* __restrict__ cnt,
                        const float* __restrict__ w_l, const float* __restrict__ b,
                        const float* __restrict__ w_r,
                        float* __restrict__ h1, int N) {
  int i = blockIdx.x * 256 + threadIdx.x;
  int node = i >> 6, o = i & 63;
  if (node >= N) return;
  float c = cnt[node];
  float inv = 1.0f / (c > 1.0f ? c : 1.0f);
  float acc = b[o];
#pragma unroll
  for (int k = 0; k < 16; ++k)
    acc += s1[node * 16 + k] * inv * w_l[o * 16 + k] + x[node * 16 + k] * w_r[o * 16 + k];
  h1[i] = acc > 0.f ? acc : 0.f;
}

// h2h[node][o] = (f16) relu(b[o] + mean . w_l + h1 . w_r),  64 -> 128
__global__ void k_node2(const float* __restrict__ h1, const float* __restrict__ s2,
                        const float* __restrict__ cnt,
                        const float* __restrict__ w_l, const float* __restrict__ b,
                        const float* __restrict__ w_r,
                        _Float16* __restrict__ h2h, int N) {
  int i = blockIdx.x * 256 + threadIdx.x;
  int node = i >> 7, o = i & 127;
  if (node >= N) return;
  float c = cnt[node];
  float inv = 1.0f / (c > 1.0f ? c : 1.0f);
  float acc = b[o];
  for (int k = 0; k < 64; ++k)
    acc += s2[node * 64 + k] * inv * w_l[o * 64 + k] + h1[node * 64 + k] * w_r[o * 64 + k];
  h2h[i] = (_Float16)(acc > 0.f ? acc : 0.f);
}

// ---------------- edge MLP (WMMA) ----------------

// One wave computes a 16-edge tile of  out[16,Nout] = relu?(in[16,K] @ W^T + b)
// entirely in LDS (in/out are this wave's private 16 rows).
__device__ __forceinline__ void wmma_layer(
    const _Float16* __restrict__ inb, int Sin,
    _Float16* __restrict__ outb, int Sout,
    const _Float16* __restrict__ frag, int KT, int NT,
    const float* __restrict__ bias, bool relu,
    int lane, int rowBase) {
  int er    = lane & 15;
  int hiofs = (lane & 16) ? 16 : 0;  // A: hi half-wave holds K+16..31
  int rofs  = (lane & 16) ? 8 : 0;  // D: hi half-wave holds rows M=8..15
  for (int nt = 0; nt < NT; ++nt) {
    float bv = bias[nt * 16 + er];  // C init = bias broadcast down column
    v8f c;
#pragma unroll
    for (int r = 0; r < 8; ++r) c[r] = bv;
    for (int kt = 0; kt < KT; ++kt) {
      U16h a, b;
      // A fragment: elements 0..15 = K kt*32+hiofs.. contiguous
      const _Float16* ap = inb + (rowBase + er) * Sin + kt * 32 + hiofs;
      a.q[0] = *(const u32x4*)(ap);
      a.q[1] = *(const u32x4*)(ap + 8);
      // B fragment: 32 contiguous bytes per lane from fragment-major weights
      const _Float16* bp = frag + ((nt * KT + kt) << 9) + (lane << 4);
      b.q[0] = *(const u32x4*)(bp);
      b.q[1] = *(const u32x4*)(bp + 8);
      c = __builtin_amdgcn_wmma_f32_16x16x32_f16(false, a.v, false, b.v,
                                                 (short)0, c, false, false);
    }
#pragma unroll
    for (int r = 0; r < 8; ++r) {
      float v = c[r];
      if (relu) v = v > 0.f ? v : 0.f;
      outb[(rowBase + rofs + r) * Sout + nt * 16 + er] = (_Float16)v;
    }
  }
  // order same-wave DS stores before the next layer's DS loads
  asm volatile("s_wait_dscnt 0" ::: "memory");
}

__launch_bounds__(128)
__global__ void k_edge_mlp(const _Float16* __restrict__ h2h,
                           const int* __restrict__ src, const int* __restrict__ dst,
                           const _Float16* __restrict__ frags,
                           const float* __restrict__ of_b1, const float* __restrict__ of_b2,
                           const float* __restrict__ of_b3, const float* __restrict__ of_b4,
                           const float* __restrict__ rf_w1, const float* __restrict__ rf_b1,
                           const float* __restrict__ rf_w2, const float* __restrict__ rf_b2,
                           const float* __restrict__ rf_w3, const float* __restrict__ rf_b3,
                           float* __restrict__ out, int E) {
  // padded row strides (264/136 halves = +16B) -> 4-bank skew per row, no DS conflicts
  __shared__ _Float16 bufA[64 * 264];
  __shared__ _Float16 bufB[64 * 136];
  int tid = threadIdx.x;
  int e0  = blockIdx.x * 64;

  // cooperative gather: feat[e] = concat(h2[src[e]], h2[dst[e]]) as f16 (512 B/edge)
  for (int i = tid; i < 64 * 32; i += 128) {
    int e = i >> 5, c = i & 31;
    int eg = e0 + e;
    if (eg < E) {
      int node = (c < 16) ? src[eg] : dst[eg];
      const u32x4* p = (const u32x4*)(h2h + (size_t)node * 128) + (c & 15);
      *(u32x4*)(&bufA[e * 264 + c * 8]) = *p;
    }
  }
  __syncthreads();

  int lane    = tid & 31;
  int rowBase = (tid >> 5) << 4;  // wave w owns edges rowBase..rowBase+15

  wmma_layer(bufA, 264, bufB, 136, frags + 0,     8, 8, of_b1, true,  lane, rowBase); // 256->128
  wmma_layer(bufB, 136, bufA, 264, frags + 32768, 4, 4, of_b2, true,  lane, rowBase); // 128->64
  wmma_layer(bufA, 264, bufB, 136, frags + 40960, 2, 2, of_b3, true,  lane, rowBase); // 64->32
  wmma_layer(bufB, 136, bufA, 264, frags + 43008, 1, 1, of_b4, false, lane, rowBase); // 32->16 (no relu)

  // relation_fc 16->8->4->50 per lane (two lanes per edge split the 50 outputs)
  int er = lane & 15;
  int e_local = rowBase + er;
  int eg = e0 + e_local;
  if (eg < E) {
    float o[16];
#pragma unroll
    for (int j = 0; j < 16; ++j) o[j] = (float)bufA[e_local * 264 + j];
    float r1[8];
    for (int i = 0; i < 8; ++i) {
      float a = rf_b1[i];
#pragma unroll
      for (int j = 0; j < 16; ++j) a += o[j] * rf_w1[i * 16 + j];
      r1[i] = a > 0.f ? a : 0.f;
    }
    float r2[4];
    for (int i = 0; i < 4; ++i) {
      float a = rf_b2[i];
#pragma unroll
      for (int j = 0; j < 8; ++j) a += r1[j] * rf_w2[i * 8 + j];
      r2[i] = a > 0.f ? a : 0.f;
    }
    int n0 = (lane < 16) ? 0 : 25;
    for (int n = n0; n < n0 + 25; ++n) {
      float a = rf_b3[n];
#pragma unroll
      for (int m = 0; m < 4; ++m) a += r2[m] * rf_w3[n * 4 + m];
      out[(size_t)eg * 50 + n] = a;
    }
  }
}

// ---------------- launch ----------------

extern "C" void kernel_launch(void* const* d_in, const int* in_sizes, int n_in,
                              void* d_out, int out_size, void* d_ws, size_t ws_size,
                              hipStream_t stream) {
  (void)n_in; (void)out_size; (void)ws_size;
  const float* x     = (const float*)d_in[0];
  const int*   eidx  = (const int*)d_in[1];
  const int*   y     = (const int*)d_in[2];
  const float* w1_l  = (const float*)d_in[3];
  const float* b1    = (const float*)d_in[4];
  const float* w1_r  = (const float*)d_in[5];
  const float* w2_l  = (const float*)d_in[6];
  const float* b2    = (const float*)d_in[7];
  const float* w2_r  = (const float*)d_in[8];
  const float* of_w1 = (const float*)d_in[9];
  const float* of_b1 = (const float*)d_in[10];
  const float* of_w2 = (const float*)d_in[11];
  const float* of_b2 = (const float*)d_in[12];
  const float* of_w3 = (const float*)d_in[13];
  const float* of_b3 = (const float*)d_in[14];
  const float* of_w4 = (const float*)d_in[15];
  const float* of_b4 = (const float*)d_in[16];
  const float* rf_w1 = (const float*)d_in[17];
  const float* rf_b1 = (const float*)d_in[18];
  const float* rf_w2 = (const float*)d_in[19];
  const float* rf_b2 = (const float*)d_in[20];
  const float* rf_w3 = (const float*)d_in[21];
  const float* rf_b3 = (const float*)d_in[22];

  const int N = in_sizes[0] / 16;
  const int E = in_sizes[2];
  const int* src = eidx;
  const int* dst = eidx + E;

  // scratch layout (fp32 region, then f16 regions), all 16B-aligned
  float* ws_f = (float*)d_ws;
  float* cnt  = ws_f;                   // [N]
  float* s1   = ws_f + (size_t)N;       // [N,16]
  float* s2   = ws_f + (size_t)17 * N;  // [N,64]
  float* h1   = ws_f + (size_t)81 * N;  // [N,64]
  _Float16* h2h   = (_Float16*)((char*)d_ws + (size_t)145 * N * 4);                    // [N,128] f16
  _Float16* frags = (_Float16*)((char*)d_ws + (size_t)145 * N * 4 + (size_t)256 * N);  // 43520 halves

  float* out_r = (float*)d_out;
  int*   out_y = (int*)(out_r + (size_t)E * 50);

  auto blk = [](long n, int t) { return (unsigned)((n + t - 1) / t); };

  long zn = (long)81 * N;  // cnt + s1 + s2 (atomic accumulators) re-zeroed every call
  k_zero_f32<<<blk(zn, 256), 256, 0, stream>>>((float*)d_ws, zn);
  k_prep_w<<<blk(43520, 256), 256, 0, stream>>>(of_w1, of_w2, of_w3, of_w4, frags);
  k_degree<<<blk(E, 256), 256, 0, stream>>>(dst, cnt, E);
  k_scatter<<<blk((long)E * 16, 256), 256, 0, stream>>>(src, dst, x, s1, 4, E);
  k_node1<<<blk((long)N * 64, 256), 256, 0, stream>>>(x, s1, cnt, w1_l, b1, w1_r, h1, N);
  k_scatter<<<blk((long)E * 64, 256), 256, 0, stream>>>(src, dst, h1, s2, 6, E);
  k_node2<<<blk((long)N * 128, 256), 256, 0, stream>>>(h1, s2, cnt, w2_l, b2, w2_r, h2h, N);
  k_edge_mlp<<<blk(E, 64), 128, 0, stream>>>(h2h, src, dst, frags,
                                             of_b1, of_b2, of_b3, of_b4,
                                             rf_w1, rf_b1, rf_w2, rf_b2, rf_w3, rf_b3,
                                             out_r, E);
  k_copy_y<<<blk(E, 256), 256, 0, stream>>>(y, out_y, E);
}
